// ExponentialConcordanceLoss_23201413333491
// MI455X (gfx1250) — compile-verified, matrix operand order checked by reading, not run on hardware
//
#include <hip/hip_runtime.h>
#include <hip/hip_bf16.h>

typedef __attribute__((ext_vector_type(16))) _Float16 v16h;
typedef __attribute__((ext_vector_type(8)))  _Float16 v8h;
typedef __attribute__((ext_vector_type(8)))  float    v8f;
typedef __attribute__((ext_vector_type(4)))  float    v4f;

#define NELEM 8192
#define KCHUNK 256
#define WAVES_PER_BLOCK 8

// ---------------------------------------------------------------------------
// Pass 1: a[i] = exp(p_i) (f32), bh[i] = f16(exp(-p_i)),
// dmask[i] = events[i]==1 ? dur[i] : +BIG (folds the event mask into the
// strict '<' duration compare; reference durations are <= 100).
// ---------------------------------------------------------------------------
__global__ void prep_kernel(const float* __restrict__ preds,
                            const float* __restrict__ targets,
                            float* __restrict__ a_arr,
                            float* __restrict__ dmask,
                            _Float16* __restrict__ bh) {
    int i = blockIdx.x * blockDim.x + threadIdx.x;
    if (i >= NELEM) return;
    float p = preds[i];
    a_arr[i] = expf(p);
    bh[i]    = (_Float16)expf(-p);
    float d = targets[2 * i];
    float e = targets[2 * i + 1];
    dmask[i] = (e == 1.0f) ? d : 3.0e38f;
}

// ---------------------------------------------------------------------------
// Pass 2: one wave owns 16 rows. Single WMMA chain:
//   A[m,k] = (dmask[k] < dur[row m]) ? exp(-p_k) : 0   (f16, built per chunk)
//   B      = loop-invariant: column 0 all-ones, rest zero (built once)
//   D[m,0] = sum_k mask * exp(-p_k)  -> row loss partial (f32 accumulate)
// Pair count done exactly with integer adds on the same compares (the 32
// lanes' A elements tile the 16x32 block exactly once).
//
// 16-bit A 16x32 layout (ISA 7.12.2): lane L holds row m = L&15; elements
// i=0..7 -> K = hi8+0..7, i=8..15 -> K = hi8+16..23, where hi8 = (L>=16)*8.
// C/D f32 16x16: VGPR r holds M = r + (L>=16 ? 8 : 0), N = L&15.
// ---------------------------------------------------------------------------
__global__ __launch_bounds__(256) void concordance_wmma_kernel(
    const float* __restrict__ targets,
    const float* __restrict__ a_arr,
    const float* __restrict__ dmask,
    const _Float16* __restrict__ bh,
    float* __restrict__ loss_part,
    float* __restrict__ cnt_part) {
    __shared__ __align__(16) float    s_dmask[KCHUNK];
    __shared__ __align__(16) _Float16 s_bh[KCHUNK];

    const int lane  = threadIdx.x & 31;
    const int wave  = threadIdx.x >> 5;
    const int gwave = blockIdx.x * WAVES_PER_BLOCK + wave;
    const int Rg    = gwave * 16;          // first global row of this wave
    const int m     = lane & 15;           // A row index / C column index
    const int hi8   = (lane >> 4) * 8;

    const float dur_row = targets[2 * (Rg + m)];

    // Loop-invariant B: column 0 = 1.0, all other columns 0.
    const _Float16 bsel = (m == 0) ? (_Float16)1.0f : (_Float16)0.0f;
    v16h B;
#pragma unroll
    for (int i = 0; i < 16; ++i) B[i] = bsel;

    const _Float16 zh = (_Float16)0.0f;
    v8f acc = {};
    int cnt = 0;

    for (int kb = 0; kb < NELEM; kb += KCHUNK) {
        s_dmask[threadIdx.x] = dmask[kb + threadIdx.x];
        s_bh[threadIdx.x]    = bh[kb + threadIdx.x];
        __syncthreads();

        for (int kk = 0; kk < KCHUNK; kk += 32) {
            const float*    dp = &s_dmask[kk + hi8];
            const _Float16* bp = &s_bh[kk + hi8];

            // Vectorized, unconditional LDS reads (all 16B aligned).
            v4f dlo0 = *(const v4f*)(dp + 0);
            v4f dlo1 = *(const v4f*)(dp + 4);
            v4f dhi0 = *(const v4f*)(dp + 16);
            v4f dhi1 = *(const v4f*)(dp + 20);
            v8h blo  = *(const v8h*)(bp + 0);
            v8h bhi  = *(const v8h*)(bp + 16);

            v16h A;
#pragma unroll
            for (int j = 0; j < 4; ++j) {
                bool c;
                c = dlo0[j] < dur_row; A[j]      = c ? blo[j]     : zh; cnt += c ? 1 : 0;
                c = dlo1[j] < dur_row; A[4 + j]  = c ? blo[4 + j] : zh; cnt += c ? 1 : 0;
                c = dhi0[j] < dur_row; A[8 + j]  = c ? bhi[j]     : zh; cnt += c ? 1 : 0;
                c = dhi1[j] < dur_row; A[12 + j] = c ? bhi[4 + j] : zh; cnt += c ? 1 : 0;
            }
            acc = __builtin_amdgcn_wmma_f32_16x16x32_f16(
                false, A, false, B, (short)0, acc, false, false);
        }
        __syncthreads();
    }

    // Epilogue: column 0 of D lives in lanes 0 (rows 0-7) and 16 (rows 8-15).
    float v_loss = 0.0f;
    if (m == 0) {
#pragma unroll
        for (int r = 0; r < 8; ++r)
            v_loss += a_arr[Rg + hi8 + r] * acc[r];
    }
    float v_cnt = (float)cnt;
#pragma unroll
    for (int off = 16; off > 0; off >>= 1) {
        v_loss += __shfl_down(v_loss, off, 32);
        v_cnt  += __shfl_down(v_cnt, off, 32);
    }
    if (lane == 0) {
        loss_part[gwave] = v_loss;
        cnt_part[gwave]  = v_cnt;
    }
}

// ---------------------------------------------------------------------------
// Pass 3: deterministic reduction of 512 wave partials -> out = loss/count.
// ---------------------------------------------------------------------------
__global__ void reduce_kernel(const float* __restrict__ loss_part,
                              const float* __restrict__ cnt_part,
                              float* __restrict__ out) {
    __shared__ float sl[512];
    __shared__ float sc[512];
    int t = threadIdx.x;
    sl[t] = loss_part[t];
    sc[t] = cnt_part[t];
    __syncthreads();
    for (int s = 256; s > 0; s >>= 1) {
        if (t < s) {
            sl[t] += sl[t + s];
            sc[t] += sc[t + s];
        }
        __syncthreads();
    }
    if (t == 0) out[0] = (sc[0] > 0.0f) ? (sl[0] / sc[0]) : 0.0f;
}

extern "C" void kernel_launch(void* const* d_in, const int* in_sizes, int n_in,
                              void* d_out, int out_size, void* d_ws, size_t ws_size,
                              hipStream_t stream) {
    (void)in_sizes; (void)n_in; (void)out_size; (void)ws_size;
    const float* preds   = (const float*)d_in[0];
    const float* targets = (const float*)d_in[1];

    char* ws = (char*)d_ws;
    float*    a_arr     = (float*)(ws);              // 8192 * 4 = 32768 B
    float*    dmask     = (float*)(ws + 32768);      // 8192 * 4 = 32768 B
    _Float16* bh        = (_Float16*)(ws + 65536);   // 8192 * 2 = 16384 B
    float*    loss_part = (float*)(ws + 81920);      //  512 * 4 =  2048 B
    float*    cnt_part  = (float*)(ws + 83968);      //  512 * 4 =  2048 B
    float*    out       = (float*)d_out;

    prep_kernel<<<NELEM / 256, 256, 0, stream>>>(preds, targets, a_arr, dmask, bh);
    concordance_wmma_kernel<<<(NELEM / 16) / WAVES_PER_BLOCK, 256, 0, stream>>>(
        targets, a_arr, dmask, bh, loss_part, cnt_part);
    reduce_kernel<<<1, 512, 0, stream>>>(loss_part, cnt_part, out);
}
